// LLAttentionBlock_77695958385386
// MI455X (gfx1250) — matrix operand order repaired and checked by reading, not compiled, not run on hardware
//
#include <hip/hip_runtime.h>
#include <hip/hip_bf16.h>

// ---------------------------------------------------------------------------
// LLAttentionBlock for MI455X (gfx1250): bf16 WMMA GEMMs + fused flash-style
// attention with the multiplicative HK gate and HV rank-1 correction.
// B=1, N=2048, D=1024, H=16, HD=64.
// ---------------------------------------------------------------------------

#define N_SEQ   2048
#define DMODEL  1024
#define NH      16
#define HDIM    64
#define SCALE_F 0.125f          // 64^-0.5
#define EPS_F   1e-6f
#define NEG_INF_F (-1e30f)

typedef __bf16 v4bf  __attribute__((ext_vector_type(4)));
typedef __bf16 v8bf  __attribute__((ext_vector_type(8)));
typedef __bf16 v16bf __attribute__((ext_vector_type(16)));
typedef float  v8f   __attribute__((ext_vector_type(8)));

// Concatenate two 8-element halves into a 16-element WMMA operand register set.
__device__ __forceinline__ v16bf cat8(v8bf lo, v8bf hi) {
  return __builtin_shufflevector(lo, hi, 0, 1, 2, 3, 4, 5, 6, 7,
                                         8, 9, 10, 11, 12, 13, 14, 15);
}

// A-fragment (16x32 bf16 MxK, ISA 7.12.2): lane row = lane&15; elements are one
// contiguous 16-wide K run, K = 16*half + j (this code's order, same as B).
// B-fragment (32x16 bf16 KxN): lane col = lane&15; elements K = 16*half + j
// (one contiguous 16-wide run).

// ---------------------------------------------------------------------------
// Generic GEMM: C[N,M] = A[N,K]*B[K,M] (+bias), f32 in/out, bf16 WMMA compute.
// Block: 256 threads (8 waves). Tile 128x64, K-step 32, double-buffered LDS
// with software pipelining: next tile's global loads are issued before the
// current tile's WMMAs, so s_wait_loadcnt lands after the math.
// B tile is stored TRANSPOSED in LDS so B-fragments are contiguous b128 loads.
// ---------------------------------------------------------------------------
__global__ __launch_bounds__(256) void gemm_bf16_wmma(
    const float* __restrict__ A, const float* __restrict__ B,
    const float* __restrict__ bias, float* __restrict__ C,
    int N, int K, int M, int has_bias)
{
  __shared__ __bf16 As[2][128][40];  // row-major, 80B rows (16B aligned)
  __shared__ __bf16 Bt[2][64][40];   // TRANSPOSED: Bt[col][k]

  const int tid  = threadIdx.x;
  const int lane = tid & 31;
  const int wave = tid >> 5;
  const int half = lane >> 4;
  const int ln   = lane & 15;
  const int row0 = blockIdx.x * 128;
  const int col0 = blockIdx.y * 64;

  float4 ra[4], rb[2];
  auto load_tile = [&](int k0) {
    #pragma unroll
    for (int i = 0; i < 4; ++i) {
      int f = tid + i * 256;                   // 0..1023 float4s of A tile
      int r = f >> 3, c4 = (f & 7) << 2;
      ra[i] = *(const float4*)&A[(size_t)(row0 + r) * K + (k0 + c4)];
    }
    #pragma unroll
    for (int i = 0; i < 2; ++i) {
      int f = tid + i * 256;                   // 0..511 float4s of B tile
      int r = f >> 4, c4 = (f & 15) << 2;
      rb[i] = *(const float4*)&B[(size_t)(k0 + r) * M + (col0 + c4)];
    }
  };
  auto store_tile = [&](int buf) {
    #pragma unroll
    for (int i = 0; i < 4; ++i) {
      int f = tid + i * 256;
      int r = f >> 3, c4 = (f & 7) << 2;
      v4bf pk = { (__bf16)ra[i].x, (__bf16)ra[i].y, (__bf16)ra[i].z, (__bf16)ra[i].w };
      *(v4bf*)&As[buf][r][c4] = pk;
    }
    #pragma unroll
    for (int i = 0; i < 2; ++i) {
      int f = tid + i * 256;
      int r = f >> 4, c4 = (f & 15) << 2;
      Bt[buf][c4 + 0][r] = (__bf16)rb[i].x;
      Bt[buf][c4 + 1][r] = (__bf16)rb[i].y;
      Bt[buf][c4 + 2][r] = (__bf16)rb[i].z;
      Bt[buf][c4 + 3][r] = (__bf16)rb[i].w;
    }
  };

  v8f acc[4] = {};
  load_tile(0);
  store_tile(0);
  int cur = 0;

  for (int k0 = 0; k0 < K; k0 += 32) {
    __syncthreads();                            // stores to buf `cur` visible
    const bool has_next = (k0 + 32 < K);
    if (has_next) load_tile(k0 + 32);           // overlap with WMMAs below

    const __bf16* arow = &As[cur][wave * 16 + ln][0];
    v16bf aF = cat8(*(const v8bf*)(arow + 16 * half),
                    *(const v8bf*)(arow + 16 * half + 8));
    v16bf bF[4];
    #pragma unroll
    for (int t = 0; t < 4; ++t) {
      const __bf16* bcol = &Bt[cur][16 * t + ln][0];
      bF[t] = cat8(*(const v8bf*)(bcol + 16 * half),
                   *(const v8bf*)(bcol + 16 * half + 8));
    }
    #pragma unroll
    for (int t = 0; t < 4; ++t)
      acc[t] = __builtin_amdgcn_wmma_f32_16x16x32_bf16(
          false, aF, false, bF[t], (short)0, acc[t], false, false);

    if (has_next) store_tile(cur ^ 1);          // waits for loads after math
    cur ^= 1;
  }

  #pragma unroll
  for (int t = 0; t < 4; ++t) {
    #pragma unroll
    for (int r = 0; r < 8; ++r) {
      int grow = row0 + wave * 16 + r + half * 8;
      int gcol = col0 + t * 16 + ln;
      float v = acc[t][r];
      if (has_bias) v += bias[gcol];
      C[(size_t)grow * M + gcol] = v;
    }
  }
}

// ---------------------------------------------------------------------------
// RoPE on Q,K (interleaved pairs, theta=10000, full head dim) and repack
// Q,K,V from [N, 3*H*HD] f32 into head-major [H][N][HD] bf16.
// One wave per (n,h); lane p owns pair (2p, 2p+1).
// ---------------------------------------------------------------------------
__global__ __launch_bounds__(32) void rope_split_kernel(
    const float* __restrict__ qkv,
    __bf16* __restrict__ Qb, __bf16* __restrict__ Kb, __bf16* __restrict__ Vb)
{
  const int n = blockIdx.x, h = blockIdx.y, p = threadIdx.x;  // p: 0..31
  float inv = __powf(10000.f, -(float)(2 * p) * (1.0f / (float)HDIM));
  float ang = (float)n * inv;
  float s, c;
  __sincosf(ang, &s, &c);

  const float* row = qkv + (size_t)n * (3 * DMODEL);
  const int off = h * HDIM + 2 * p;
  float q0 = row[off],              q1 = row[off + 1];
  float k0 = row[DMODEL + off],     k1 = row[DMODEL + off + 1];
  float v0 = row[2 * DMODEL + off], v1 = row[2 * DMODEL + off + 1];

  size_t o = ((size_t)h * N_SEQ + n) * HDIM + 2 * p;
  Qb[o]     = (__bf16)(q0 * c - q1 * s);
  Qb[o + 1] = (__bf16)(q1 * c + q0 * s);
  Kb[o]     = (__bf16)(k0 * c - k1 * s);
  Kb[o + 1] = (__bf16)(k1 * c + k0 * s);
  Vb[o]     = (__bf16)v0;
  Vb[o + 1] = (__bf16)v1;
}

// ---------------------------------------------------------------------------
// RMSNorm of HK, HV per (n, head) row of 64; bf16 for WMMA operands, f32 HV
// for the rank-1 output correction. One wave per (n,h).
// ---------------------------------------------------------------------------
__global__ __launch_bounds__(32) void rmsnorm_split_kernel(
    const float* __restrict__ hkv,
    const float* __restrict__ hk_w, const float* __restrict__ hv_w,
    __bf16* __restrict__ HKb, __bf16* __restrict__ HVb, float* __restrict__ HVf)
{
  const int n = blockIdx.x, h = blockIdx.y, p = threadIdx.x;
  const float* row = hkv + (size_t)n * (2 * DMODEL);
  const int off = h * HDIM + 2 * p;
  float a0 = row[off],          a1 = row[off + 1];
  float b0 = row[DMODEL + off], b1 = row[DMODEL + off + 1];

  float ssA = a0 * a0 + a1 * a1;
  float ssB = b0 * b0 + b1 * b1;
  #pragma unroll
  for (int d = 1; d < 32; d <<= 1) {
    ssA += __shfl_xor(ssA, d, 32);
    ssB += __shfl_xor(ssB, d, 32);
  }
  float rA = rsqrtf(ssA * (1.0f / (float)HDIM) + EPS_F);
  float rB = rsqrtf(ssB * (1.0f / (float)HDIM) + EPS_F);

  size_t o = ((size_t)h * N_SEQ + n) * HDIM + 2 * p;
  float hk0 = a0 * rA * hk_w[2 * p], hk1 = a1 * rA * hk_w[2 * p + 1];
  float hv0 = b0 * rB * hv_w[2 * p], hv1 = b1 * rB * hv_w[2 * p + 1];
  HKb[o] = (__bf16)hk0; HKb[o + 1] = (__bf16)hk1;
  HVb[o] = (__bf16)hv0; HVb[o + 1] = (__bf16)hv1;
  HVf[o] = hv0;         HVf[o + 1] = hv1;
}

// ---------------------------------------------------------------------------
// Fused causal attention with gate + rank-1 correction.
// Block = 128 threads = 4 independent waves; wave w owns query tile
// qt = 4*blockIdx.x + w (16 rows) of head blockIdx.y. 32 keys per iteration:
//   S = Q*K^T, T = HK*K^T, U = HV*V^T  (2 key subtiles x 2 K-chunks WMMA each)
//   score = (1 + beta*relu(T)) * S * SCALE, causal mask, online softmax with
//   running m/l and rr = sum(P*U). The V rows loaded for U are transposed
//   through a private LDS tile (Vt[dim][key]) so the P@V B-fragments are
//   contiguous ds_load_b128, and P is routed through LDS into A-fragment form.
// No block barriers: LDS ops are in-order per wave; slices are per-wave.
// Epilogue: out = O/l + alpha * (rr/l) * HV.
// ---------------------------------------------------------------------------
__global__ __launch_bounds__(128) void attn_fused_kernel(
    const __bf16* __restrict__ Qb, const __bf16* __restrict__ Kb,
    const __bf16* __restrict__ Vb, const __bf16* __restrict__ HKb,
    const __bf16* __restrict__ HVb, const float* __restrict__ HVf,
    const float* __restrict__ alpha, const float* __restrict__ beta,
    float* __restrict__ Out)
{
  __shared__ __bf16 Pl[4][16][40];   // P tile per wave (16 queries x 32 keys)
  __shared__ __bf16 Vl[4][64][40];   // transposed V tile per wave: Vt[dim][key]

  const int wave = threadIdx.x >> 5;
  const int lane = threadIdx.x & 31;
  const int half = lane >> 4, ln = lane & 15;
  const int qt = blockIdx.x * 4 + wave;
  const int h  = blockIdx.y;
  const int q0 = qt * 16;
  __bf16 (*P)[40]  = Pl[wave];
  __bf16 (*VT)[40] = Vl[wave];

  const __bf16* Qh  = Qb  + (size_t)h * N_SEQ * HDIM;
  const __bf16* Kh  = Kb  + (size_t)h * N_SEQ * HDIM;
  const __bf16* Vh  = Vb  + (size_t)h * N_SEQ * HDIM;
  const __bf16* HKh = HKb + (size_t)h * N_SEQ * HDIM;
  const __bf16* HVh = HVb + (size_t)h * N_SEQ * HDIM;
  const float*  HVfh = HVf + (size_t)h * N_SEQ * HDIM;
  const float alphah = alpha[h];
  const float betah  = beta[h];

  // A-fragments for Q, HK, HV: two contiguous v8bf runs per 32-wide K chunk.
  v16bf aQ[2], aHK[2], aHV[2];
  {
    const size_t rbase = (size_t)(q0 + ln) * HDIM;
    #pragma unroll
    for (int c = 0; c < 2; ++c) {
      const size_t lo = rbase + 32 * c + 16 * half;
      aQ[c]  = cat8(*(const v8bf*)(Qh + lo),  *(const v8bf*)(Qh + lo + 8));
      aHK[c] = cat8(*(const v8bf*)(HKh + lo), *(const v8bf*)(HKh + lo + 8));
      aHV[c] = cat8(*(const v8bf*)(HVh + lo), *(const v8bf*)(HVh + lo + 8));
    }
  }

  float mrow[8], lrow[8], rrow[8];
  #pragma unroll
  for (int r = 0; r < 8; ++r) { mrow[r] = -3.0e38f; lrow[r] = 0.f; rrow[r] = 0.f; }
  v8f accO[4] = {};

  const int kend = q0 + 16;                  // keys needed: [0, kend)
  for (int k0 = 0; k0 < kend; k0 += 32) {
    v8f S[2] = {}, T[2] = {}, U[2] = {};
    #pragma unroll
    for (int s = 0; s < 2; ++s) {
      const size_t krow = (size_t)(k0 + 16 * s + ln) * HDIM;
      #pragma unroll
      for (int c = 0; c < 2; ++c) {
        // B-fragment: contiguous 16 bf16 (32B aligned) of the key's row.
        const size_t idx = krow + 32 * c + 16 * half;
        v16bf bK  = *(const v16bf*)(Kh + idx);
        v16bf bVT = *(const v16bf*)(Vh + idx);
        S[s] = __builtin_amdgcn_wmma_f32_16x16x32_bf16(false, aQ[c],  false, bK,  (short)0, S[s], false, false);
        T[s] = __builtin_amdgcn_wmma_f32_16x16x32_bf16(false, aHK[c], false, bK,  (short)0, T[s], false, false);
        U[s] = __builtin_amdgcn_wmma_f32_16x16x32_bf16(false, aHV[c], false, bVT, (short)0, U[s], false, false);
        // Stage the V rows (already in registers) transposed for the P@V step:
        // VT[dim][key]; one base address + immediate offsets.
        #pragma unroll
        for (int j = 0; j < 16; ++j)
          VT[32 * c + 16 * half + j][16 * s + ln] = bVT[j];
      }
    }

    // Gate, mask, online softmax. C-layout: elem r of lane = (row r+8*half, col ln).
    #pragma unroll
    for (int r = 0; r < 8; ++r) {
      const int qg = q0 + r + 8 * half;
      float sc0 = (1.f + betah * fmaxf(T[0][r], 0.f)) * S[0][r] * SCALE_F;
      float sc1 = (1.f + betah * fmaxf(T[1][r], 0.f)) * S[1][r] * SCALE_F;
      if (k0 + ln > qg)      sc0 = NEG_INF_F;
      if (k0 + 16 + ln > qg) sc1 = NEG_INF_F;

      float tmax = fmaxf(sc0, sc1);
      #pragma unroll
      for (int d = 1; d < 16; d <<= 1) tmax = fmaxf(tmax, __shfl_xor(tmax, d, 32));
      float mnew = fmaxf(mrow[r], tmax);
      float corr = __expf(mrow[r] - mnew);
      mrow[r] = mnew;

      float e0 = __expf(sc0 - mnew);
      float e1 = __expf(sc1 - mnew);
      float psum = e0 + e1;
      float rsum = e0 * U[0][r] + e1 * U[1][r];
      #pragma unroll
      for (int d = 1; d < 16; d <<= 1) {
        psum += __shfl_xor(psum, d, 32);
        rsum += __shfl_xor(rsum, d, 32);
      }
      lrow[r] = lrow[r] * corr + psum;
      rrow[r] = rrow[r] * corr + rsum;
      #pragma unroll
      for (int t = 0; t < 4; ++t) accO[t][r] *= corr;

      P[r + 8 * half][ln]      = (__bf16)e0;
      P[r + 8 * half][16 + ln] = (__bf16)e1;
    }

    // P (16x32) as A-fragment; V^T columns as contiguous B-fragments from LDS.
    const __bf16* prow = &P[ln][0];
    v16bf aP = cat8(*(const v8bf*)(prow + 16 * half),
                    *(const v8bf*)(prow + 16 * half + 8));
    #pragma unroll
    for (int t = 0; t < 4; ++t) {
      const __bf16* vcol = &VT[16 * t + ln][0];
      v16bf bVo = cat8(*(const v8bf*)(vcol + 16 * half),
                       *(const v8bf*)(vcol + 16 * half + 8));
      accO[t] = __builtin_amdgcn_wmma_f32_16x16x32_bf16(
          false, aP, false, bVo, (short)0, accO[t], false, false);
    }
  }

  // Epilogue: normalize and add alpha * (rr/l) * HV; write [N, H*HD].
  #pragma unroll
  for (int r = 0; r < 8; ++r) {
    const int row = q0 + r + 8 * half;
    const float invl = 1.f / lrow[r];
    const float rterm = alphah * rrow[r] * invl;
    #pragma unroll
    for (int t = 0; t < 4; ++t) {
      const int d = 16 * t + ln;
      float v = accO[t][r] * invl + rterm * HVfh[(size_t)row * HDIM + d];
      Out[(size_t)row * (NH * HDIM) + h * HDIM + d] = v;
    }
  }
}

// ---------------------------------------------------------------------------
// Launch
// ---------------------------------------------------------------------------
extern "C" void kernel_launch(void* const* d_in, const int* in_sizes, int n_in,
                              void* d_out, int out_size, void* d_ws, size_t ws_size,
                              hipStream_t stream) {
  (void)in_sizes; (void)n_in; (void)out_size; (void)ws_size;
  const float* X     = (const float*)d_in[0];
  const float* W_qkv = (const float*)d_in[1];
  const float* W_hkv = (const float*)d_in[2];
  const float* W_out = (const float*)d_in[3];
  const float* b_out = (const float*)d_in[4];
  const float* alpha = (const float*)d_in[5];
  const float* beta  = (const float*)d_in[6];
  const float* hk_w  = (const float*)d_in[7];
  const float* hv_w  = (const float*)d_in[8];
  float* out = (float*)d_out;

  char* ws = (char*)d_ws;
  size_t off = 0;
  auto alloc = [&](size_t bytes) -> char* {
    char* p = ws + off;
    off += (bytes + 255) & ~(size_t)255;
    return p;
  };
  float*  qkv  = (float*) alloc((size_t)N_SEQ * 3 * DMODEL * sizeof(float));
  float*  hkv  = (float*) alloc((size_t)N_SEQ * 2 * DMODEL * sizeof(float));
  __bf16* Qb   = (__bf16*)alloc((size_t)NH * N_SEQ * HDIM * sizeof(__bf16));
  __bf16* Kb   = (__bf16*)alloc((size_t)NH * N_SEQ * HDIM * sizeof(__bf16));
  __bf16* Vb   = (__bf16*)alloc((size_t)NH * N_SEQ * HDIM * sizeof(__bf16));
  __bf16* HKb  = (__bf16*)alloc((size_t)NH * N_SEQ * HDIM * sizeof(__bf16));
  __bf16* HVb  = (__bf16*)alloc((size_t)NH * N_SEQ * HDIM * sizeof(__bf16));
  float*  HVf  = (float*) alloc((size_t)NH * N_SEQ * HDIM * sizeof(float));
  float*  attn = (float*) alloc((size_t)N_SEQ * NH * HDIM * sizeof(float));

  const dim3 blk256(256);
  // 1-2) Projections: X @ W_qkv -> [N,3072], X @ W_hkv -> [N,2048]
  gemm_bf16_wmma<<<dim3(N_SEQ / 128, (3 * DMODEL) / 64), blk256, 0, stream>>>(
      X, W_qkv, nullptr, qkv, N_SEQ, DMODEL, 3 * DMODEL, 0);
  gemm_bf16_wmma<<<dim3(N_SEQ / 128, (2 * DMODEL) / 64), blk256, 0, stream>>>(
      X, W_hkv, nullptr, hkv, N_SEQ, DMODEL, 2 * DMODEL, 0);
  // 3) RoPE(Q,K) + repack V to head-major bf16
  rope_split_kernel<<<dim3(N_SEQ, NH), dim3(32), 0, stream>>>(qkv, Qb, Kb, Vb);
  // 4) RMSNorm(HK,HV)
  rmsnorm_split_kernel<<<dim3(N_SEQ, NH), dim3(32), 0, stream>>>(
      hkv, hk_w, hv_w, HKb, HVb, HVf);
  // 5) Fused attention (4 waves per block, one query tile each)
  attn_fused_kernel<<<dim3(N_SEQ / 64, NH), dim3(128), 0, stream>>>(
      Qb, Kb, Vb, HKb, HVb, HVf, alpha, beta, attn);
  // 6) Output projection with bias
  gemm_bf16_wmma<<<dim3(N_SEQ / 128, DMODEL / 64), blk256, 0, stream>>>(
      attn, W_out, b_out, out, N_SEQ, DMODEL, DMODEL, 1);
}
